// MRFFI_43963285242156
// MI455X (gfx1250) — matrix operand order repaired and checked by reading, not compiled, not run on hardware
//
#include <hip/hip_runtime.h>
#include <hip/hip_bf16.h>
#include <hip/hip_fp16.h>

// ---------------- problem constants ----------------
#define BATCH   8
#define HH      64
#define WW      64
#define LSEQ    (HH*WW)          // 4096 per batch
#define MROWS   (BATCH*LSEQ)     // 32768
#define DIM     320
#define DG      192
#define DL      96
#define D_INNER 384
#define D_STATE 16
#define DT_RANK 12
#define XDBL    44               // dt_rank + 2*d_state
#define XDBL_P  48               // padded to 3*16
#define EPSV    1e-5f

typedef __attribute__((ext_vector_type(16))) _Float16 v16h;
typedef __attribute__((ext_vector_type(8)))  _Float16 v8h;
typedef __attribute__((ext_vector_type(8)))  float    v8f;
typedef __attribute__((ext_vector_type(4)))  int      v4i;

typedef __attribute__((address_space(1))) v4i* gas_v4i_ptr;
typedef __attribute__((address_space(3))) v4i* las_v4i_ptr;

union V16U { v16h v; v8h h[2]; };

// ---- gfx1250 async copy global->LDS (ASYNCcnt-tracked), with safe fallback ----
__device__ __forceinline__ void async_copy16(const _Float16* g, _Float16* l) {
#if __has_builtin(__builtin_amdgcn_global_load_async_to_lds_b128)
    __builtin_amdgcn_global_load_async_to_lds_b128(
        (gas_v4i_ptr)(uintptr_t)g, (las_v4i_ptr)(uintptr_t)l, 0, 0);
#else
    *(v8h*)l = *(const v8h*)g;
#endif
}

__device__ __forceinline__ void wait_async_copies() {
#if __has_builtin(__builtin_amdgcn_global_load_async_to_lds_b128)
#if __has_builtin(__builtin_amdgcn_s_wait_asynccnt)
    __builtin_amdgcn_s_wait_asynccnt(0);
#else
    asm volatile("s_wait_asynccnt 0x0" ::: "memory");
#endif
#endif
}

// ---- WMMA fragment loaders from LDS (CDNA5 16x16x32 f16 layouts, ISA 7.12.2) ----
// A (16x32 MxK): lanes 0-15: M=lane, K=k0+{0..7,16..23}; lanes 16-31: M=lane-16, K=k0+{8..15,24..31}
__device__ __forceinline__ v16h frag_a(const _Float16* p, int ldp, int row0, int k0, int lane) {
    int r    = row0 + (lane & 15);
    int koff = k0 + ((lane >> 4) << 3);
    const _Float16* base = p + (size_t)r * ldp + koff;
    V16U f;
    f.h[0] = *(const v8h*)(base);
    f.h[1] = *(const v8h*)(base + 16);
    return f.v;
}

// B (32x16 KxN) from row-major W (NxK): lane holds column n=n0+(lane&15);
// lanes 0-15: K=k0+{0..7,16..23}; lanes 16-31: K=k0+{8..15,24..31}
__device__ __forceinline__ v16h frag_b(const _Float16* p, int ldp, int n0, int k0, int lane) {
    int n    = n0 + (lane & 15);
    int koff = k0 + ((lane >> 4) << 3);
    const _Float16* base = p + (size_t)n * ldp + koff;
    V16U f;
    f.h[0] = *(const v8h*)(base);
    f.h[1] = *(const v8h*)(base + 16);
    return f.v;
}

// ---------------- LDS-staged WMMA GEMM: C[M,N] = A[M,K] * W[N,K]^T (+bias) ----------------
// A, W must be densely packed (lda = ldb = K). Block: 256 threads = 8 waves.
// Block tile: 128 rows x (NT*16) cols. Panels staged to LDS via async copies.
template<int NT, int K>
__global__ void __launch_bounds__(256) wmma_gemm_kernel(
        const _Float16* __restrict__ A, const _Float16* __restrict__ W,
        float* __restrict__ C, int ldc, const float* __restrict__ bias, int Nreal) {
    constexpr int KP = K + 8;            // padded row (halves): keeps 16B align, skews banks
    constexpr int NB = NT * 16;
    __shared__ __align__(16) _Float16 As[128 * KP];
    __shared__ __align__(16) _Float16 Bs[NB * KP];

    const int tid = threadIdx.x;
    // stage A panel (128 x K, contiguous at A + blockIdx.x*128*K)
    {
        const _Float16* gA = A + (size_t)blockIdx.x * 128 * K;
        constexpr int CH = 128 * K / 8;            // 16B chunks
        for (int i = tid; i < CH; i += 256) {
            int r = i / (K / 8), c = i % (K / 8);
            async_copy16(gA + (size_t)r * K + c * 8, &As[r * KP + c * 8]);
        }
    }
    // stage B panel (NB x K, contiguous at W + blockIdx.y*NB*K)
    {
        const _Float16* gB = W + (size_t)blockIdx.y * NB * K;
        constexpr int CH = NB * K / 8;
        for (int i = tid; i < CH; i += 256) {
            int r = i / (K / 8), c = i % (K / 8);
            async_copy16(gB + (size_t)r * K + c * 8, &Bs[r * KP + c * 8]);
        }
    }
    wait_async_copies();
    __syncthreads();

    const int lane = tid & 31;
    const int wid  = tid >> 5;
    const int rowl = wid * 16;

    v8f acc[NT];
#pragma unroll
    for (int t = 0; t < NT; ++t)
#pragma unroll
        for (int r = 0; r < 8; ++r) acc[t][r] = 0.0f;

#pragma unroll 2
    for (int k0 = 0; k0 < K; k0 += 32) {
        v16h a = frag_a(As, KP, rowl, k0, lane);
#pragma unroll
        for (int t = 0; t < NT; ++t) {
            v16h b = frag_b(Bs, KP, t * 16, k0, lane);
            acc[t] = __builtin_amdgcn_wmma_f32_16x16x32_f16(
                false, a, false, b, (short)0, acc[t], false, false);
        }
    }

    // C/D layout: lanes 0-15: VGPR r -> M=row+r, N=lane; lanes 16-31: M=row+8+r, N=lane-16.
    const int mrow = blockIdx.x * 128 + rowl + ((lane >> 4) << 3);
    const int ncl  = lane & 15;
#pragma unroll
    for (int t = 0; t < NT; ++t) {
        int n = blockIdx.y * NB + t * 16 + ncl;
        if (n < Nreal) {
            float bv = bias ? bias[n] : 0.0f;
#pragma unroll
            for (int r = 0; r < 8; ++r)
                C[(size_t)(mrow + r) * ldc + n] = acc[t][r] + bv;
        }
    }
}

// ---------------- prep kernels ----------------
__global__ void cvt_f32_to_f16_kernel(const float* __restrict__ s, _Float16* __restrict__ d, int n) {
    int i = blockIdx.x * blockDim.x + threadIdx.x;
    if (i < n) d[i] = (_Float16)s[i];
}

__global__ void pad_xproj_kernel(const float* __restrict__ s, _Float16* __restrict__ d) {
    int i = blockIdx.x * blockDim.x + threadIdx.x;
    if (i >= XDBL_P * D_INNER) return;
    int n = i / D_INNER, k = i % D_INNER;
    d[i] = (n < XDBL) ? (_Float16)s[n * D_INNER + k] : (_Float16)0.0f;
}

__global__ void neg_exp_kernel(const float* __restrict__ a, float* __restrict__ o, int n) {
    int i = blockIdx.x * blockDim.x + threadIdx.x;
    if (i < n) o[i] = -__expf(a[i]);
}

__global__ void zero_f32_kernel(float* p, int n) {
    int i = blockIdx.x * blockDim.x + threadIdx.x;
    if (i < n) p[i] = 0.0f;
}

// ---------------- LayerNorm over first DG channels, one wave per row ----------------
__global__ void layernorm_kernel(const float* __restrict__ x, const float* __restrict__ g,
                                 const float* __restrict__ bt, _Float16* __restrict__ out) {
    int wid  = (blockIdx.x * blockDim.x + threadIdx.x) >> 5;
    int lane = threadIdx.x & 31;
    if (wid >= MROWS) return;
    const float* row = x + (size_t)wid * DIM;
    float v[6];
    float sum = 0.0f;
#pragma unroll
    for (int i = 0; i < 6; ++i) { v[i] = row[lane + i * 32]; sum += v[i]; }
#pragma unroll
    for (int o = 16; o > 0; o >>= 1) sum += __shfl_xor(sum, o, 32);
    float mu = sum * (1.0f / DG);
    float var = 0.0f;
#pragma unroll
    for (int i = 0; i < 6; ++i) { float d = v[i] - mu; var += d * d; }
#pragma unroll
    for (int o = 16; o > 0; o >>= 1) var += __shfl_xor(var, o, 32);
    float inv = rsqrtf(var * (1.0f / DG) + EPSV);
#pragma unroll
    for (int i = 0; i < 6; ++i) {
        int c = lane + i * 32;
        out[(size_t)wid * DG + c] = (_Float16)((v[i] - mu) * inv * g[c] + bt[c]);
    }
}

// ---------------- causal depthwise conv1d (k=4) + SiLU ----------------
__global__ void conv1d_silu_kernel(const float* __restrict__ xz, const float* __restrict__ w,
                                   const float* __restrict__ b, _Float16* __restrict__ u_h,
                                   float* __restrict__ u_f) {
    int idx = blockIdx.x * blockDim.x + threadIdx.x;
    if (idx >= MROWS * D_INNER) return;
    int c = idx % D_INNER;
    int m = idx / D_INNER;
    int l = m & (LSEQ - 1);
    float acc = b[c];
#pragma unroll
    for (int j = 0; j < 4; ++j) {
        int lj = l - 3 + j;
        if (lj >= 0) acc += w[c * 4 + j] * xz[(size_t)(m - 3 + j) * (2 * D_INNER) + c];
    }
    float sv = acc / (1.0f + __expf(-acc));
    u_h[idx] = (_Float16)sv;
    u_f[idx] = sv;
}

// ---------------- dt = softplus(dt_r @ dt_proj_w^T + b) ----------------
__global__ void dt_kernel(const float* __restrict__ xdbl, const float* __restrict__ w,
                          const float* __restrict__ b, float* __restrict__ dt) {
    int idx = blockIdx.x * blockDim.x + threadIdx.x;
    if (idx >= MROWS * D_INNER) return;
    int c = idx % D_INNER;
    int m = idx / D_INNER;
    float a = b[c];
    const float* xr = xdbl + (size_t)m * XDBL_P;
#pragma unroll
    for (int r = 0; r < DT_RANK; ++r) a += xr[r] * w[c * DT_RANK + r];
    dt[idx] = (a > 20.0f) ? a : log1pf(__expf(a));
}

// ---------------- selective scan: one lane per (batch, channel) ----------------
__global__ void scan_kernel(const float* __restrict__ xz, const float* __restrict__ xdbl,
                            const float* __restrict__ dtb, const float* __restrict__ u_f,
                            const float* __restrict__ Aneg, const float* __restrict__ Dp,
                            _Float16* __restrict__ ygate) {
    int tid = blockIdx.x * blockDim.x + threadIdx.x;
    if (tid >= BATCH * D_INNER) return;
    int c = tid % D_INNER, b = tid / D_INNER;
    float A[D_STATE], h[D_STATE];
#pragma unroll
    for (int s = 0; s < D_STATE; ++s) { A[s] = Aneg[c * D_STATE + s]; h[s] = 0.0f; }
    float Dv = Dp[c];
    size_t mb = (size_t)b * LSEQ;
    for (int l = 0; l < LSEQ; ++l) {
        size_t m   = mb + l;
        float dtv  = dtb[m * D_INNER + c];
        float uv   = u_f[m * D_INNER + c];
        const float* xd = xdbl + m * XDBL_P;
        float du = dtv * uv;
        float y = 0.0f;
#pragma unroll
        for (int s = 0; s < D_STATE; ++s) {
            float dA = __expf(dtv * A[s]);
            h[s] = h[s] * dA + du * xd[DT_RANK + s];
            y += h[s] * xd[DT_RANK + D_STATE + s];
        }
        y += uv * Dv;
        float z = xz[m * (2 * D_INNER) + D_INNER + c];
        float g = z / (1.0f + __expf(-z));
        ygate[m * D_INNER + c] = (_Float16)(y * g);
    }
}

// ---------------- depthwise 3/5/7 convs on local branch ----------------
__global__ void dwconv_kernel(const float* __restrict__ x, const float* __restrict__ w3,
                              const float* __restrict__ w5, const float* __restrict__ w7,
                              float* __restrict__ outs) {
    const size_t NP = (size_t)BATCH * DL * LSEQ;
    int idx = blockIdx.x * blockDim.x + threadIdx.x;
    if (idx >= (int)NP) return;
    int hw = idx & (LSEQ - 1);
    int c  = (idx / LSEQ) % DL;
    int b  = idx / (LSEQ * DL);
    int yy = hw >> 6, xx = hw & 63;
    float o3 = 0.0f, o5 = 0.0f, o7 = 0.0f;
    for (int dy = -3; dy <= 3; ++dy) {
        int py = yy + dy;
        for (int dx = -3; dx <= 3; ++dx) {
            int px = xx + dx;
            float v = (py >= 0 && py < HH && px >= 0 && px < WW)
                          ? x[(((size_t)b * HH + py) * WW + px) * DIM + DG + c] : 0.0f;
            o7 += v * w7[c * 49 + (dy + 3) * 7 + (dx + 3)];
            if (dy >= -2 && dy <= 2 && dx >= -2 && dx <= 2)
                o5 += v * w5[c * 25 + (dy + 2) * 5 + (dx + 2)];
            if (dy >= -1 && dy <= 1 && dx >= -1 && dx <= 1)
                o3 += v * w3[c * 9 + (dy + 1) * 3 + (dx + 1)];
        }
    }
    size_t plane = ((size_t)b * DL + c) * LSEQ + hw;
    outs[plane] = o3; outs[NP + plane] = o5; outs[2 * NP + plane] = o7;
}

// ---------------- s[b,c] = mean over HW of (o3+o5+o7) ----------------
__global__ void s_reduce_kernel(const float* __restrict__ outs, float* __restrict__ s) {
    __shared__ float rs[256];
    const size_t NP = (size_t)BATCH * DL * LSEQ;
    int bc = blockIdx.x;
    size_t plane = (size_t)bc * LSEQ;
    float acc = 0.0f;
    for (int i = threadIdx.x; i < LSEQ; i += 256)
        acc += outs[plane + i] + outs[NP + plane + i] + outs[2 * NP + plane + i];
    rs[threadIdx.x] = acc; __syncthreads();
    for (int st = 128; st > 0; st >>= 1) {
        if ((int)threadIdx.x < st) rs[threadIdx.x] += rs[threadIdx.x + st];
        __syncthreads();
    }
    if (threadIdx.x == 0) s[bc] = rs[0] * (1.0f / LSEQ);
}

// ---------------- SK attention: relu(s@fc1^T)@fc2^T, softmax over 3 per channel ----------------
__global__ void att_kernel(const float* __restrict__ s, const float* __restrict__ fc1,
                           const float* __restrict__ fc2, float* __restrict__ att) {
    __shared__ float sv[DL], hid[24], lg[3 * DL];
    int b = blockIdx.x, t = threadIdx.x;   // 288 threads
    if (t < DL) sv[t] = s[b * DL + t];
    __syncthreads();
    if (t < 24) {
        float a = 0.0f;
        for (int k = 0; k < DL; ++k) a += fc1[t * DL + k] * sv[k];
        hid[t] = fmaxf(a, 0.0f);
    }
    __syncthreads();
    float logit = 0.0f;
#pragma unroll
    for (int k = 0; k < 24; ++k) logit += fc2[t * 24 + k] * hid[k];
    lg[t] = logit; __syncthreads();
    int c = t / 3, k = t % 3;
    float l0 = lg[c * 3], l1 = lg[c * 3 + 1], l2 = lg[c * 3 + 2];
    float mx = fmaxf(l0, fmaxf(l1, l2));
    float e  = __expf(lg[t] - mx);
    float sm = __expf(l0 - mx) + __expf(l1 - mx) + __expf(l2 - mx);
    att[b * (3 * DL) + k * DL + c] = e / sm;   // stored as (b, branch, channel)
}

// ---------------- yl = sum_k att*outs ; accumulate BN stats ----------------
__global__ void yl_bn_stats_kernel(const float* __restrict__ outs, const float* __restrict__ att,
                                   float* __restrict__ yl, float* bnsum, float* bnsq) {
    __shared__ float rs[256], rq[256];
    const size_t NP = (size_t)BATCH * DL * LSEQ;
    int bc = blockIdx.x;
    int b = bc / DL, c = bc % DL;
    float a3 = att[b * (3 * DL) + 0 * DL + c];
    float a5 = att[b * (3 * DL) + 1 * DL + c];
    float a7 = att[b * (3 * DL) + 2 * DL + c];
    size_t plane = (size_t)bc * LSEQ;
    float s = 0.0f, q = 0.0f;
    for (int i = threadIdx.x; i < LSEQ; i += 256) {
        float v = a3 * outs[plane + i] + a5 * outs[NP + plane + i] + a7 * outs[2 * NP + plane + i];
        yl[plane + i] = v; s += v; q += v * v;
    }
    rs[threadIdx.x] = s; rq[threadIdx.x] = q; __syncthreads();
    for (int st = 128; st > 0; st >>= 1) {
        if ((int)threadIdx.x < st) {
            rs[threadIdx.x] += rs[threadIdx.x + st];
            rq[threadIdx.x] += rq[threadIdx.x + st];
        }
        __syncthreads();
    }
    if (threadIdx.x == 0) { atomicAdd(&bnsum[c], rs[0]); atomicAdd(&bnsq[c], rq[0]); }
}

__global__ void bn_finalize_kernel(const float* bnsum, const float* bnsq,
                                   float* mean, float* inv) {
    int c = threadIdx.x;
    if (c >= DL) return;
    const float n = (float)(BATCH * LSEQ);
    float mu = bnsum[c] / n;
    float var = bnsq[c] / n - mu * mu;
    mean[c] = mu;
    inv[c]  = rsqrtf(var + EPSV);
}

// ---------------- build f16 concat [yg | bn(yl) | x_i] ----------------
__global__ void concat_kernel(const float* __restrict__ yg, const float* __restrict__ yl,
                              const float* __restrict__ mean, const float* __restrict__ inv,
                              const float* __restrict__ bw, const float* __restrict__ bb,
                              const float* __restrict__ x, _Float16* __restrict__ cc) {
    int idx = blockIdx.x * blockDim.x + threadIdx.x;
    if (idx >= MROWS * DIM) return;
    int col = idx % DIM, m = idx / DIM;
    _Float16 v;
    if (col < DG) {
        v = (_Float16)yg[(size_t)m * DG + col];
    } else if (col < DG + DL) {
        int c = col - DG;
        int b = m >> 12, hw = m & (LSEQ - 1);
        float val = yl[((size_t)b * DL + c) * LSEQ + hw];
        v = (_Float16)((val - mean[c]) * inv[c] * bw[c] + bb[c]);
    } else {
        v = (_Float16)x[(size_t)m * DIM + col];
    }
    cc[idx] = v;
}

// ---------------- host side ----------------
static inline int cdiv(int a, int b) { return (a + b - 1) / b; }

extern "C" void kernel_launch(void* const* d_in, const int* in_sizes, int n_in,
                              void* d_out, int out_size, void* d_ws, size_t ws_size,
                              hipStream_t stream) {
    const float* x         = (const float*)d_in[0];
    const float* ln_g_w    = (const float*)d_in[1];
    const float* ln_g_b    = (const float*)d_in[2];
    const float* in_proj_w = (const float*)d_in[3];
    const float* conv1d_w  = (const float*)d_in[4];
    const float* conv1d_b  = (const float*)d_in[5];
    const float* x_proj_w  = (const float*)d_in[6];
    const float* dt_proj_w = (const float*)d_in[7];
    const float* dt_proj_b = (const float*)d_in[8];
    const float* A_log     = (const float*)d_in[9];
    const float* Dp        = (const float*)d_in[10];
    const float* out_proj_w= (const float*)d_in[11];
    const float* conv3_w   = (const float*)d_in[12];
    const float* conv5_w   = (const float*)d_in[13];
    const float* conv7_w   = (const float*)d_in[14];
    const float* fc1_w     = (const float*)d_in[15];
    const float* fc2_w     = (const float*)d_in[16];
    const float* bn_w      = (const float*)d_in[17];
    const float* bn_b      = (const float*)d_in[18];
    const float* proj_w    = (const float*)d_in[19];
    const float* proj_b    = (const float*)d_in[20];
    float* out = (float*)d_out;

    // workspace carve-up
    char* p = (char*)d_ws;
    auto take = [&](size_t bytes) { void* r = (void*)p; p += (bytes + 255) & ~(size_t)255; return r; };
    _Float16* xg_h    = (_Float16*)take((size_t)MROWS * DG * 2);
    _Float16* w_in_h  = (_Float16*)take((size_t)2 * D_INNER * DG * 2);
    _Float16* w_xp_h  = (_Float16*)take((size_t)XDBL_P * D_INNER * 2);
    _Float16* w_out_h = (_Float16*)take((size_t)DG * D_INNER * 2);
    _Float16* w_pj_h  = (_Float16*)take((size_t)DIM * DIM * 2);
    float*    Aneg    = (float*)take((size_t)D_INNER * D_STATE * 4);
    float*    xz      = (float*)take((size_t)MROWS * 2 * D_INNER * 4);
    _Float16* u_h     = (_Float16*)take((size_t)MROWS * D_INNER * 2);
    float*    u_f     = (float*)take((size_t)MROWS * D_INNER * 4);
    float*    xdbl    = (float*)take((size_t)MROWS * XDBL_P * 4);
    float*    dtb     = (float*)take((size_t)MROWS * D_INNER * 4);
    _Float16* ygate   = (_Float16*)take((size_t)MROWS * D_INNER * 2);
    float*    yg      = (float*)take((size_t)MROWS * DG * 4);
    float*    outs    = (float*)take((size_t)3 * BATCH * DL * LSEQ * 4);
    float*    sbuf    = (float*)take((size_t)BATCH * DL * 4);
    float*    attb    = (float*)take((size_t)BATCH * 3 * DL * 4);
    float*    ylb     = (float*)take((size_t)BATCH * DL * LSEQ * 4);
    float*    bnsum   = (float*)take(DL * 4);
    float*    bnsq    = (float*)take(DL * 4);
    float*    bnmean  = (float*)take(DL * 4);
    float*    bninv   = (float*)take(DL * 4);
    _Float16* concat  = (_Float16*)take((size_t)MROWS * DIM * 2);

    // ---- prep: weight conversions, A = -exp(A_log), zero BN accumulators ----
    zero_f32_kernel<<<1, 256, 0, stream>>>(bnsum, DL);
    zero_f32_kernel<<<1, 256, 0, stream>>>(bnsq, DL);
    {
        int n1 = 2 * D_INNER * DG;
        cvt_f32_to_f16_kernel<<<cdiv(n1, 256), 256, 0, stream>>>(in_proj_w, w_in_h, n1);
        int n2 = DG * D_INNER;
        cvt_f32_to_f16_kernel<<<cdiv(n2, 256), 256, 0, stream>>>(out_proj_w, w_out_h, n2);
        int n3 = DIM * DIM;
        cvt_f32_to_f16_kernel<<<cdiv(n3, 256), 256, 0, stream>>>(proj_w, w_pj_h, n3);
        int n4 = XDBL_P * D_INNER;
        pad_xproj_kernel<<<cdiv(n4, 256), 256, 0, stream>>>(x_proj_w, w_xp_h);
        int n5 = D_INNER * D_STATE;
        neg_exp_kernel<<<cdiv(n5, 256), 256, 0, stream>>>(A_log, Aneg, n5);
    }

    // ---- Mamba chain ----
    layernorm_kernel<<<MROWS / 8, 256, 0, stream>>>(x, ln_g_w, ln_g_b, xg_h);

    // xz = xg @ in_proj_w^T : M=32768, N=768, K=192
    {
        dim3 grid(MROWS / 128, (2 * D_INNER) / 64);
        wmma_gemm_kernel<4, DG><<<grid, 256, 0, stream>>>(xg_h, w_in_h, xz, 2 * D_INNER,
                                                          nullptr, 2 * D_INNER);
    }

    conv1d_silu_kernel<<<cdiv(MROWS * D_INNER, 256), 256, 0, stream>>>(xz, conv1d_w, conv1d_b, u_h, u_f);

    // xdbl = u @ x_proj_w^T : N padded to 48 (NT=3), guard Nreal=44
    {
        dim3 grid(MROWS / 128, 1);
        wmma_gemm_kernel<3, D_INNER><<<grid, 256, 0, stream>>>(u_h, w_xp_h, xdbl, XDBL_P,
                                                               nullptr, XDBL);
    }

    dt_kernel<<<cdiv(MROWS * D_INNER, 256), 256, 0, stream>>>(xdbl, dt_proj_w, dt_proj_b, dtb);

    scan_kernel<<<cdiv(BATCH * D_INNER, 256), 256, 0, stream>>>(xz, xdbl, dtb, u_f, Aneg, Dp, ygate);

    // yg = ygate @ out_proj_w^T : M=32768, N=192, K=384
    {
        dim3 grid(MROWS / 128, DG / 64);
        wmma_gemm_kernel<4, D_INNER><<<grid, 256, 0, stream>>>(ygate, w_out_h, yg, DG,
                                                               nullptr, DG);
    }

    // ---- local branch ----
    dwconv_kernel<<<cdiv(BATCH * DL * LSEQ, 256), 256, 0, stream>>>(x, conv3_w, conv5_w, conv7_w, outs);
    s_reduce_kernel<<<BATCH * DL, 256, 0, stream>>>(outs, sbuf);
    att_kernel<<<BATCH, 3 * DL, 0, stream>>>(sbuf, fc1_w, fc2_w, attb);
    yl_bn_stats_kernel<<<BATCH * DL, 256, 0, stream>>>(outs, attb, ylb, bnsum, bnsq);
    bn_finalize_kernel<<<1, 128, 0, stream>>>(bnsum, bnsq, bnmean, bninv);

    // ---- concat + final projection ----
    concat_kernel<<<cdiv(MROWS * DIM, 256), 256, 0, stream>>>(yg, ylb, bnmean, bninv, bn_w, bn_b,
                                                              x, concat);
    {
        dim3 grid(MROWS / 128, DIM / 64);
        wmma_gemm_kernel<4, DIM><<<grid, 256, 0, stream>>>(concat, w_pj_h, out, DIM,
                                                           proj_b, DIM);
    }
}
